// GraphLayer_87582973100245
// MI455X (gfx1250) — compile-verified, matrix operand order checked by reading, not run on hardware
//
#include <hip/hip_runtime.h>

typedef __bf16        v16bf __attribute__((ext_vector_type(16)));
typedef float         v8f   __attribute__((ext_vector_type(8)));
typedef unsigned int  v8u   __attribute__((ext_vector_type(8)));

#define N_NODES 8192
#define D_IN    512
#define D_OUT   512   // HEADS * DPH = 8*64

static __device__ __forceinline__ unsigned fbits(float x) {
    return __builtin_bit_cast(unsigned, x);
}

// Round-to-nearest-even f32->bf16 pair pack (used only in the small stage-1).
static __device__ __forceinline__ unsigned pack2_bf16_rne(float a, float b) {
    unsigned ua = fbits(a), ub = fbits(b);
    ua += 0x7FFFu + ((ua >> 16) & 1u);
    ub += 0x7FFFu + ((ub >> 16) & 1u);
    return (ua >> 16) | (ub & 0xFFFF0000u);
}

// Truncating f32->bf16 pair pack: ONE v_perm_b32. dest = {hi16(b), hi16(a)}.
// Exact for 0.0/1.0 adjacency values.
static __device__ __forceinline__ unsigned pack2_bf16_trunc(float a, float b) {
    return __builtin_amdgcn_perm(fbits(b), fbits(a), 0x07060302u);
}

// Load one 16x32 (MxK) A-fragment for v_wmma_f32_16x16x32_bf16 from a
// row-major f32 matrix.
// ISA layout: lanes 0-15 hold M=lane, K = kk+[0..7] and kk+[16..23];
// lanes 16-31 hold M=lane-16, K = kk+[8..15] and kk+[24..31].
template <bool RNE>
static __device__ __forceinline__ v16bf load_a_frag_f32(
    const float* __restrict__ base, long ldm, int m0, int kk, int lane) {
    const int row  = m0 + (lane & 15);
    const int koff = (lane < 16) ? 0 : 8;
    const float* p = base + (long)row * ldm + kk + koff;
    float4 r0a = *(const float4*)(p + 0);
    float4 r0b = *(const float4*)(p + 4);
    float4 r1a = *(const float4*)(p + 16);
    float4 r1b = *(const float4*)(p + 20);
    v8u pk;
    if (RNE) {
        pk[0] = pack2_bf16_rne(r0a.x, r0a.y);
        pk[1] = pack2_bf16_rne(r0a.z, r0a.w);
        pk[2] = pack2_bf16_rne(r0b.x, r0b.y);
        pk[3] = pack2_bf16_rne(r0b.z, r0b.w);
        pk[4] = pack2_bf16_rne(r1a.x, r1a.y);
        pk[5] = pack2_bf16_rne(r1a.z, r1a.w);
        pk[6] = pack2_bf16_rne(r1b.x, r1b.y);
        pk[7] = pack2_bf16_rne(r1b.z, r1b.w);
    } else {
        pk[0] = pack2_bf16_trunc(r0a.x, r0a.y);
        pk[1] = pack2_bf16_trunc(r0a.z, r0a.w);
        pk[2] = pack2_bf16_trunc(r0b.x, r0b.y);
        pk[3] = pack2_bf16_trunc(r0b.z, r0b.w);
        pk[4] = pack2_bf16_trunc(r1a.x, r1a.y);
        pk[5] = pack2_bf16_trunc(r1a.z, r1a.w);
        pk[6] = pack2_bf16_trunc(r1b.x, r1b.y);
        pk[7] = pack2_bf16_trunc(r1b.z, r1b.w);
    }
    return __builtin_bit_cast(v16bf, pk);
}

// ---------------------------------------------------------------------------
// Stage 1: Hp(col-major, D_OUT x N, bf16) = bf16(H @ Wc), Wc = head-concat(W)
// One wave -> one 16x16 tile, K = 512 in 16 WMMA steps. (~6% of total FLOPs)
// ---------------------------------------------------------------------------
__global__ __launch_bounds__(256) void proj_kernel(
    const float* __restrict__ H, const float* __restrict__ W,
    __bf16* __restrict__ Hp) {
    const int lane = threadIdx.x & 31;
    const int wave = threadIdx.x >> 5;
    const int m0 = blockIdx.y * 128 + wave * 16;   // node rows
    const int n0 = blockIdx.x * 16;                // Hp columns

    // B-fragment column: Wc element (d, col) = W[h*512*64 + d*64 + e]
    const int col  = n0 + (lane & 15);
    const int head = col >> 6;
    const int e    = col & 63;
    const float* wp = W + (long)head * (D_IN * 64) + e;
    const int koff = (lane < 16) ? 0 : 8;

    v8f acc = {};
    for (int kk = 0; kk < D_IN; kk += 32) {
        v16bf a = load_a_frag_f32<true>(H, D_IN, m0, kk, lane);
        // B fragment 32x16 (KxN): lane holds fixed col, K runs mirror A layout.
        const float* q0 = wp + (long)(kk + koff) * 64;
        const float* q1 = wp + (long)(kk + 16 + koff) * 64;
        v8u pk;
#pragma unroll
        for (int j = 0; j < 4; ++j) {
            pk[j]     = pack2_bf16_rne(q0[(2 * j) * 64], q0[(2 * j + 1) * 64]);
            pk[4 + j] = pack2_bf16_rne(q1[(2 * j) * 64], q1[(2 * j + 1) * 64]);
        }
        v16bf b = __builtin_bit_cast(v16bf, pk);
        acc = __builtin_amdgcn_wmma_f32_16x16x32_bf16(
            false, a, false, b, (short)0, acc, false, false);
    }

    // C layout: VGPR i -> M = m0 + (lane<16 ? 0 : 8) + i, N = col.
    // Column-major store: 8 consecutive rows = one aligned 16B bf16 store.
    const int rowbase = m0 + ((lane >> 4) << 3);
    uint4 st;
    st.x = pack2_bf16_rne(acc[0], acc[1]);
    st.y = pack2_bf16_rne(acc[2], acc[3]);
    st.z = pack2_bf16_rne(acc[4], acc[5]);
    st.w = pack2_bf16_rne(acc[6], acc[7]);
    *(uint4*)(Hp + (long)col * N_NODES + rowbase) = st;
}

// ---------------------------------------------------------------------------
// Stage 2: OUT(N x D_OUT, f32) = adj @ Hp + b_flat  (68.7 GFLOP, 256 MB adj)
// Block = 8 waves, block tile 128x128; wave tile 32x64:
//   2 A-frags x 4 B-frags -> 8 WMMAs per K-step against ~24 other issue slots,
//   so the XDL pipes stay fed. adj f32->bf16 via one v_perm_b32 per pair
//   (exact for {0,1}).
// ---------------------------------------------------------------------------
__global__ __launch_bounds__(256) void aggr_kernel(
    const float* __restrict__ adj, const __bf16* __restrict__ Hp,
    const float* __restrict__ bias, float* __restrict__ out) {
    const int lane = threadIdx.x & 31;
    const int wave = threadIdx.x >> 5;
    const int wm = wave & 3;                       // 4 M sub-tiles of 32 rows
    const int wn = wave >> 2;                      // 2 N sub-tiles of 64 cols
    const int m0 = blockIdx.y * 128 + wm * 32;
    const int n0 = blockIdx.x * 128 + wn * 64;

    const int koff = (lane < 16) ? 0 : 8;
    // 4 column-group base pointers into column-major bf16 Hp.
    const __bf16* hp[4];
#pragma unroll
    for (int f = 0; f < 4; ++f)
        hp[f] = Hp + (long)(n0 + 16 * f + (lane & 15)) * N_NODES + koff;
    const float* arow = adj + (long)(m0 + (lane & 15)) * N_NODES;

    v8f acc0[4] = {{}, {}, {}, {}};   // rows m0..m0+15
    v8f acc1[4] = {{}, {}, {}, {}};   // rows m0+16..m0+31

    for (int kk = 0; kk < N_NODES; kk += 32) {
        // Branch-free prefetch ~16 K-steps ahead (wraps within the row).
        __builtin_prefetch(arow + ((kk + 512) & (N_NODES - 1)), 0, 1);

        v16bf a0 = load_a_frag_f32<false>(adj, N_NODES, m0,      kk, lane);
        v16bf a1 = load_a_frag_f32<false>(adj, N_NODES, m0 + 16, kk, lane);

        v16bf b[4];
#pragma unroll
        for (int f = 0; f < 4; ++f) {
            uint4 r0 = *(const uint4*)(hp[f] + kk);
            uint4 r1 = *(const uint4*)(hp[f] + kk + 16);
            v8u pb = {r0.x, r0.y, r0.z, r0.w, r1.x, r1.y, r1.z, r1.w};
            b[f] = __builtin_bit_cast(v16bf, pb);
        }

#pragma unroll
        for (int f = 0; f < 4; ++f) {
            acc0[f] = __builtin_amdgcn_wmma_f32_16x16x32_bf16(
                false, a0, false, b[f], (short)0, acc0[f], false, false);
            acc1[f] = __builtin_amdgcn_wmma_f32_16x16x32_bf16(
                false, a1, false, b[f], (short)0, acc1[f], false, false);
        }
    }

    // Epilogue: add bias (flat index == output column), store f32 row-major.
    const int r0 = m0 + ((lane >> 4) << 3);
    const int r1 = r0 + 16;
#pragma unroll
    for (int f = 0; f < 4; ++f) {
        const int col = n0 + 16 * f + (lane & 15);
        const float bv = bias[col];
#pragma unroll
        for (int i = 0; i < 8; ++i) {
            out[(long)(r0 + i) * D_OUT + col] = acc0[f][i] + bv;
            out[(long)(r1 + i) * D_OUT + col] = acc1[f][i] + bv;
        }
    }
}

extern "C" void kernel_launch(void* const* d_in, const int* in_sizes, int n_in,
                              void* d_out, int out_size, void* d_ws, size_t ws_size,
                              hipStream_t stream) {
    const float* H   = (const float*)d_in[0];   // (8192, 512)
    const float* adj = (const float*)d_in[1];   // (8192, 8192)
    const float* W   = (const float*)d_in[2];   // (8, 512, 64)
    const float* b   = (const float*)d_in[3];   // (8, 64) -> flat == column
    float* out = (float*)d_out;                 // (8192, 512)

    __bf16* Hp = (__bf16*)d_ws;                 // column-major 512 x 8192 bf16, 8 MB

    dim3 blk(256);
    dim3 g1(D_OUT / 16, N_NODES / 128);         // 32 x 64 blocks, 8 waves each
    proj_kernel<<<g1, blk, 0, stream>>>(H, W, Hp);

    dim3 g2(D_OUT / 128, N_NODES / 128);        // 4 x 64 blocks
    aggr_kernel<<<g2, blk, 0, stream>>>(adj, Hp, b, out);
}
